// LSTMCell_76398878261544
// MI455X (gfx1250) — matrix operand order repaired and checked by reading, not compiled, not run on hardware
//
#include <hip/hip_runtime.h>
#include <stdint.h>

// ---------------------------------------------------------------------------
// LSTM cell, B=8192, I=H=1024, fp32 in/out.
// Strategy: fp32 -> bf16 convert pass into d_ws, then fused 4-gate GEMM with
// v_wmma_f32_16x16x32_bf16 + async global->LDS double buffering + fused
// sigmoid/tanh epilogue writing ht, ct directly.
// d_ws usage: 32MB A=[x|h] bf16 (8192x2048) + 4 * 4MB W=[Wi|Wh] bf16 = 48MB.
// ---------------------------------------------------------------------------

#define BATCH 8192
#define HID   1024
#define KDIM  2048      // fused K = I + H
#define BM    128
#define BN    64
#define BK    32
#define KSTEPS (KDIM / BK)          // 64
#define A_LDS_STRIDE 40             // 32 + 8 pad (bf16 elems) -> conflict-free
#define A_LDS_ELEMS (BM * A_LDS_STRIDE)           // 5120
#define W_LDS_ELEMS (4 * BN * A_LDS_STRIDE)       // 10240
#define STAGE_ELEMS (A_LDS_ELEMS + W_LDS_ELEMS)   // 15360 elems = 30720 B

typedef __attribute__((ext_vector_type(16))) __bf16 v16bf;
typedef __attribute__((ext_vector_type(8)))  float  v8f;

union Frag32B { v16bf v; uint4 q[2]; };

// ---------------- fp32 -> bf16 (RNE) convert pass --------------------------

__device__ __forceinline__ uint32_t f2bf(float f) {
    uint32_t u = __float_as_uint(f);
    u += 0x7fffu + ((u >> 16) & 1u);   // round-to-nearest-even
    return u >> 16;
}

// dst is [rows][2048] bf16; dst[:, 0:1024] = cvt(s0), dst[:, 1024:2048] = cvt(s1)
__global__ void __launch_bounds__(256) cvt_pair_kernel(
    const float* __restrict__ s0, const float* __restrict__ s1,
    uint16_t* __restrict__ dst, int rows)
{
    int idx = blockIdx.x * blockDim.x + threadIdx.x;   // one 4-float chunk
    int total = rows * (HID / 4);
    if (idx >= total) return;
    int r = idx >> 8;              // HID/4 = 256 chunks per row
    int col = (idx & 255) * 4;

    float4 a = *reinterpret_cast<const float4*>(s0 + r * HID + col);
    float4 b = *reinterpret_cast<const float4*>(s1 + r * HID + col);
    uint2 pa, pb;
    pa.x = f2bf(a.x) | (f2bf(a.y) << 16);
    pa.y = f2bf(a.z) | (f2bf(a.w) << 16);
    pb.x = f2bf(b.x) | (f2bf(b.y) << 16);
    pb.y = f2bf(b.z) | (f2bf(b.w) << 16);
    *reinterpret_cast<uint2*>(dst + r * KDIM + col)       = pa;
    *reinterpret_cast<uint2*>(dst + r * KDIM + HID + col) = pb;
}

// ---------------- fused 4-gate GEMM + LSTM epilogue ------------------------

__device__ __forceinline__ void async_ld16(uint32_t lds_byte, const void* g) {
    asm volatile("global_load_async_to_lds_b128 %0, %1, off"
                 :: "v"(lds_byte), "v"((uint64_t)(uintptr_t)g) : "memory");
}

__device__ __forceinline__ float sigm(float x) {
    return 1.0f / (1.0f + __expf(-x));
}
__device__ __forceinline__ float tanh_fast(float x) {
    x = fminf(fmaxf(x, -15.0f), 15.0f);
    float e = __expf(-2.0f * x);
    return (1.0f - e) / (1.0f + e);
}

__global__ void __launch_bounds__(256) lstm_gemm_kernel(
    const uint16_t* __restrict__ Abf,     // [8192][2048] bf16
    const uint16_t* __restrict__ Wbf,     // 4 x [1024][2048] bf16, gate-major
    const float* __restrict__ bf_i, const float* __restrict__ bf_h,
    const float* __restrict__ bi_i, const float* __restrict__ bi_h,
    const float* __restrict__ bg_i, const float* __restrict__ bg_h,
    const float* __restrict__ bo_i, const float* __restrict__ bo_h,
    const float* __restrict__ c_in,
    float* __restrict__ out_h, float* __restrict__ out_c)
{
    __shared__ __align__(16) uint16_t smem[2][STAGE_ELEMS];

    const int tid  = threadIdx.x;
    const int lane = tid & 31;
    const int wid  = tid >> 5;
    const int waveM = wid >> 1;          // 0..3 -> 32-row slab
    const int waveN = wid & 1;           // 0..1 -> 32-col slab
    const int m0 = blockIdx.y * BM;
    const int n0 = blockIdx.x * BN;

    const uint32_t lds_base = (uint32_t)(uintptr_t)(&smem[0][0]);

    // ---- per-thread async copy descriptors (A: 2 chunks, W: 4 chunks) ----
    // A chunk id c in [0,512): row=c>>2, seg=c&3 (16B = 8 bf16 per seg)
    // W chunk id c in [0,1024): gate=c>>8, r=(c>>2)&63, seg=c&3
    int a_row[2], a_seg[2];
    {
        int c0 = tid, c1 = tid + 256;
        a_row[0] = c0 >> 2; a_seg[0] = c0 & 3;
        a_row[1] = c1 >> 2; a_seg[1] = c1 & 3;
    }
    int w_g[4], w_r[4], w_seg[4];
    for (int j = 0; j < 4; ++j) {
        int c = tid + j * 256;
        w_g[j] = c >> 8; w_r[j] = (c >> 2) & 63; w_seg[j] = c & 3;
    }

    auto issue_stage = [&](int buf, int kt) {
        const int kc = kt * BK;
        const uint32_t sbase = lds_base + (uint32_t)(buf * STAGE_ELEMS * 2);
        #pragma unroll
        for (int j = 0; j < 2; ++j) {
            const uint16_t* g = Abf + (size_t)(m0 + a_row[j]) * KDIM + kc + a_seg[j] * 8;
            uint32_t l = sbase + (uint32_t)(a_row[j] * A_LDS_STRIDE + a_seg[j] * 8) * 2;
            async_ld16(l, g);
        }
        #pragma unroll
        for (int j = 0; j < 4; ++j) {
            const uint16_t* g = Wbf + (size_t)w_g[j] * (HID * KDIM)
                              + (size_t)(n0 + w_r[j]) * KDIM + kc + w_seg[j] * 8;
            uint32_t l = sbase + (uint32_t)(A_LDS_ELEMS
                              + (w_g[j] * BN + w_r[j]) * A_LDS_STRIDE + w_seg[j] * 8) * 2;
            async_ld16(l, g);
        }
    };

    v8f acc[4][2][2];
    const v8f vz = {0.f, 0.f, 0.f, 0.f, 0.f, 0.f, 0.f, 0.f};
    #pragma unroll
    for (int g = 0; g < 4; ++g)
        #pragma unroll
        for (int mi = 0; mi < 2; ++mi)
            #pragma unroll
            for (int ni = 0; ni < 2; ++ni)
                acc[g][mi][ni] = vz;

    issue_stage(0, 0);

    const int khalf = (lane >> 4) * 8;      // A frag: k split by lane half
    const int kseg  = (lane >> 4) * 8;      // B frag: k split by lane half

    #pragma unroll 1
    for (int kt = 0; kt < KSTEPS; ++kt) {
        asm volatile("s_wait_asynccnt 0" ::: "memory");
        __syncthreads();
        if (kt + 1 < KSTEPS) issue_stage((kt + 1) & 1, kt + 1);

        const uint16_t* As = &smem[kt & 1][0];
        const uint16_t* Ws = As + A_LDS_ELEMS;

        // A fragments: rows (waveM*32 + mi*16 + lane%16)
        Frag32B afr[2];
        #pragma unroll
        for (int mi = 0; mi < 2; ++mi) {
            int m = waveM * 32 + mi * 16 + (lane & 15);
            const uint16_t* p = As + m * A_LDS_STRIDE + khalf;
            afr[mi].q[0] = *reinterpret_cast<const uint4*>(p);
            afr[mi].q[1] = *reinterpret_cast<const uint4*>(p + 16);
        }

        #pragma unroll
        for (int g = 0; g < 4; ++g) {
            Frag32B bfr[2];
            #pragma unroll
            for (int ni = 0; ni < 2; ++ni) {
                int n = waveN * 32 + ni * 16 + (lane & 15);
                const uint16_t* p = Ws + (g * BN + n) * A_LDS_STRIDE + kseg;
                bfr[ni].q[0] = *reinterpret_cast<const uint4*>(p);
                bfr[ni].q[1] = *reinterpret_cast<const uint4*>(p + 16);
            }
            #pragma unroll
            for (int mi = 0; mi < 2; ++mi)
                #pragma unroll
                for (int ni = 0; ni < 2; ++ni)
                    acc[g][mi][ni] = __builtin_amdgcn_wmma_f32_16x16x32_bf16(
                        false, afr[mi].v, false, bfr[ni].v,
                        (short)0, acc[g][mi][ni], false, false);
        }
        __syncthreads();
    }

    // ------------------------------ epilogue ------------------------------
    const float* biasI[4] = { bf_i, bi_i, bg_i, bo_i };
    const float* biasH[4] = { bf_h, bi_h, bg_h, bo_h };

    #pragma unroll
    for (int ni = 0; ni < 2; ++ni) {
        const int n = n0 + waveN * 32 + ni * 16 + (lane & 15);
        float bsum[4];
        #pragma unroll
        for (int g = 0; g < 4; ++g) bsum[g] = biasI[g][n] + biasH[g][n];

        #pragma unroll
        for (int mi = 0; mi < 2; ++mi) {
            const int mbase = m0 + waveM * 32 + mi * 16 + 8 * (lane >> 4);
            #pragma unroll
            for (int r = 0; r < 8; ++r) {
                const int m = mbase + r;
                float pf = acc[0][mi][ni][r] + bsum[0];
                float pi = acc[1][mi][ni][r] + bsum[1];
                float pg = acc[2][mi][ni][r] + bsum[2];
                float po = acc[3][mi][ni][r] + bsum[3];
                float ft = sigm(pf);
                float it = sigm(pi);
                float gt = tanh_fast(pg);
                float ot = sigm(po);
                float cp = c_in[(size_t)m * HID + n];
                float ct = ft * cp + it * gt;
                float ht = ot * tanh_fast(ct);
                out_h[(size_t)m * HID + n] = ht;
                out_c[(size_t)m * HID + n] = ct;
            }
        }
    }
}

// ---------------------------------------------------------------------------

extern "C" void kernel_launch(void* const* d_in, const int* in_sizes, int n_in,
                              void* d_out, int out_size, void* d_ws, size_t ws_size,
                              hipStream_t stream) {
    const float* x  = (const float*)d_in[0];
    const float* h  = (const float*)d_in[1];
    const float* c  = (const float*)d_in[2];
    const float* wf_i = (const float*)d_in[3];  const float* bf_i = (const float*)d_in[4];
    const float* wf_h = (const float*)d_in[5];  const float* bf_h = (const float*)d_in[6];
    const float* wi_i = (const float*)d_in[7];  const float* bi_i = (const float*)d_in[8];
    const float* wi_h = (const float*)d_in[9];  const float* bi_h = (const float*)d_in[10];
    const float* wg_i = (const float*)d_in[11]; const float* bg_i = (const float*)d_in[12];
    const float* wg_h = (const float*)d_in[13]; const float* bg_h = (const float*)d_in[14];
    const float* wo_i = (const float*)d_in[15]; const float* bo_i = (const float*)d_in[16];
    const float* wo_h = (const float*)d_in[17]; const float* bo_h = (const float*)d_in[18];

    float* out_h = (float*)d_out;                      // ht: 8192*1024
    float* out_c = out_h + (size_t)BATCH * HID;        // ct: 8192*1024

    // workspace: A bf16 then 4 gate weight matrices bf16 (f, i, g, o)
    uint16_t* Abf = (uint16_t*)d_ws;                           // 32 MB
    uint16_t* Wbf = Abf + (size_t)BATCH * KDIM;                // 16 MB

    // --- convert pass ---
    {
        int chunks = BATCH * (HID / 4);
        cvt_pair_kernel<<<(chunks + 255) / 256, 256, 0, stream>>>(x, h, Abf, BATCH);
        int wchunks = HID * (HID / 4);
        dim3 g((wchunks + 255) / 256);
        cvt_pair_kernel<<<g, 256, 0, stream>>>(wf_i, wf_h, Wbf + 0 * (size_t)HID * KDIM, HID);
        cvt_pair_kernel<<<g, 256, 0, stream>>>(wi_i, wi_h, Wbf + 1 * (size_t)HID * KDIM, HID);
        cvt_pair_kernel<<<g, 256, 0, stream>>>(wg_i, wg_h, Wbf + 2 * (size_t)HID * KDIM, HID);
        cvt_pair_kernel<<<g, 256, 0, stream>>>(wo_i, wo_h, Wbf + 3 * (size_t)HID * KDIM, HID);
    }

    // --- fused GEMM + LSTM epilogue ---
    dim3 grid(HID / BN, BATCH / BM);   // 16 x 64
    lstm_gemm_kernel<<<grid, 256, 0, stream>>>(
        Abf, Wbf,
        bf_i, bf_h, bi_i, bi_h, bg_i, bg_h, bo_i, bo_h,
        c, out_h, out_c);
}
